// ThirteenBranch_53206054863478
// MI455X (gfx1250) — matrix operand order repaired and checked by reading, not compiled, not run on hardware
//
#include <hip/hip_runtime.h>

// ---- problem constants ----
#define BATCH 16
#define CINCH 2
#define DSZ   64
#define COUT  13
#define KK    5
#define NROW  50          // r = (cin,kd,kh) halo rows
#define RPAD  64          // rows per kw-block (K slot: kw = k>>6, r = k&63)
#define POSN  68          // 64 + 4 x-halo positions
#define STR   72          // shorts per pos row: 144B (16B-aligned, bank-spread 36 dw)
#define NSTEP 10          // K = 5*64 = 320 = 10 x K32 WMMA steps
#define WAVES 4           // 128 threads / block

typedef __attribute__((ext_vector_type(16))) __bf16 v16bf;
typedef __attribute__((ext_vector_type(8)))  float  v8f;
typedef __attribute__((ext_vector_type(4)))  unsigned int v4u;

union Frag16 { v16bf v; v4u q[2]; unsigned short h[16]; };

__device__ __forceinline__ unsigned short f2bf(float f) {
    return __builtin_bit_cast(unsigned short, (__bf16)f);   // native v_cvt, RNE
}

__global__ __launch_bounds__(WAVES * 32)
void conv3d13_wmma_kernel(const float* __restrict__ x,
                          const float* __restrict__ W,
                          const float* __restrict__ bias,
                          float* __restrict__ out)
{
    // per-wave: transposed bf16 halo stage [pos][r], and f32 out-transpose buffer
    __shared__ unsigned short s_in[WAVES][POSN * STR];
    __shared__ float          s_out[WAVES][16 * DSZ];

    const int tid  = threadIdx.x;
    const int wave = tid >> 5;
    const int lane = tid & 31;

    // ---- which output row (b, z, y) this wave owns ----
    const int row = blockIdx.x * WAVES + wave;
    const int b = row >> 12;
    const int z = (row >> 6) & 63;
    const int y = row & 63;

    // ---- stage x halo into LDS transposed: lds[pos][r] = x[..., pos-2] (bf16) ----
    unsigned short* sin = s_in[wave];
    #pragma unroll 1
    for (int r = 0; r < NROW; ++r) {
        int cin = r / 25, rem = r - cin * 25;
        int kd  = rem / 5, kh  = rem - kd * 5;
        int zz = z + kd - 2;
        int yy = y + kh - 2;
        bool rowok = ((unsigned)zz < DSZ) & ((unsigned)yy < DSZ);
        const float* src = x + ((((size_t)(b * CINCH + cin) * DSZ + zz) * DSZ + yy) * DSZ);
        #pragma unroll
        for (int j = 0; j < 3; ++j) {
            int pos = lane + j * 32;
            if (pos < POSN) {
                int xm = pos - 2;
                float v = (rowok && (unsigned)xm < DSZ) ? src[xm] : 0.0f;
                sin[pos * STR + r] = f2bf(v);
            }
        }
    }
    // zero rows r = 50..63 (K padding) for every pos; r=50 is 4B-aligned
    #pragma unroll 1
    for (int j = 0; j < 3; ++j) {
        int pos = lane + j * 32;
        if (pos < POSN) {
            unsigned* p = (unsigned*)(sin + pos * STR + NROW);
            #pragma unroll
            for (int q = 0; q < (RPAD - NROW) / 2; ++q) p[q] = 0u;
        }
    }

    // ---- B fragments in VGPRs (10 steps): B[k][n], kw-major K mapping ----
    // lanes 0-15: N=lane, h[i] = K = t*32 + (i&7) + (i>>3)*16 ; lanes 16-31: same +8
    Frag16 bfr[NSTEP];
    {
        const int n   = lane & 15;
        const int kof = (lane >> 4) << 3;              // 0 or 8
        #pragma unroll
        for (int t = 0; t < NSTEP; ++t) {
            #pragma unroll
            for (int i = 0; i < 16; ++i) {
                int k  = t * 32 + kof + (i & 7) + ((i >> 3) << 4);
                int kw = k >> 6, r = k & 63;           // kw-major K slot
                float wv = (n < COUT && r < NROW) ? W[n * (NROW * KK) + r * KK + kw] : 0.0f;
                bfr[t].h[i] = f2bf(wv);
            }
        }
    }

    // ---- C init with bias ----
    float bv; { int n = lane & 15; bv = (n < COUT) ? bias[n] : 0.0f; }
    v8f c[4];
    #pragma unroll
    for (int tle = 0; tle < 4; ++tle)
        #pragma unroll
        for (int v = 0; v < 8; ++v) c[tle][v] = bv;

    // ---- main loop: 10 K-steps x 4 M-tiles; A half-fragment = 1 ds_load_b128 ----
    // A: lanes 0-15 (koA=0) M=lane ; lanes 16-31 (koA=8) M=lane-16
    //    h[0..7] = K = t*32+koA+0..7 ; h[8..15] = K = t*32+koA+16..23
    const int m   = lane & 15;
    const int koA = (lane >> 4) << 3;                  // 0 or 8, shifts r within row
    const unsigned short* ap = sin + m * STR + koA;    // single gather base address
    #pragma unroll
    for (int t = 0; t < NSTEP; ++t) {
        #pragma unroll
        for (int tle = 0; tle < 4; ++tle) {
            const int k0 = t * 32;                     // compile-time per site
            const int k1 = t * 32 + 16;
            const int off0 = tle * 16 * STR + (k0 >> 6) * STR + (k0 & 63);
            const int off1 = tle * 16 * STR + (k1 >> 6) * STR + (k1 & 63);
            Frag16 a;
            a.q[0] = *(const v4u*)(ap + off0);         // ds_load_b128, 16B aligned
            a.q[1] = *(const v4u*)(ap + off1);         // ds_load_b128
            c[tle] = __builtin_amdgcn_wmma_f32_16x16x32_bf16(
                false, a.v, false, bfr[t].v, (short)0, c[tle], false, false);
        }
    }

    // ---- epilogue: transpose C through LDS for coalesced f32 stores ----
    float* so = s_out[wave];
    {
        const int n    = lane & 15;
        const int mofs = (lane >> 4) << 3;
        #pragma unroll
        for (int tle = 0; tle < 4; ++tle)
            #pragma unroll
            for (int v = 0; v < 8; ++v)
                so[n * DSZ + (tle * 16 + mofs + v)] = c[tle][v];
    }
    // same-wave LDS is in-order; DScnt waits inserted by compiler.

    #pragma unroll 1
    for (int n = 0; n < COUT; ++n) {
        float* dn = out + ((((size_t)(b * COUT + n)) * DSZ + z) * DSZ + y) * DSZ;
        #pragma unroll
        for (int j = 0; j < 2; ++j) {
            int xx = lane + j * 32;
            float v = so[n * DSZ + xx];
            v = fminf(fmaxf(v, 0.0f), 100.0f);
            dn[xx] = v;                                // lane-contiguous 128B bursts
        }
    }
}

extern "C" void kernel_launch(void* const* d_in, const int* in_sizes, int n_in,
                              void* d_out, int out_size, void* d_ws, size_t ws_size,
                              hipStream_t stream) {
    (void)in_sizes; (void)n_in; (void)d_ws; (void)ws_size; (void)out_size;
    const float* x    = (const float*)d_in[0];
    const float* W    = (const float*)d_in[1];
    const float* bias = (const float*)d_in[2];
    float* out        = (float*)d_out;

    const int rows   = BATCH * DSZ * DSZ;              // 65536 (b,z,y) rows
    const int blocks = rows / WAVES;                   // 16384
    conv3d13_wmma_kernel<<<blocks, WAVES * 32, 0, stream>>>(x, W, bias, out);
}
